// Attention_60249801228320
// MI455X (gfx1250) — matrix operand order repaired and checked by reading, not compiled, not run on hardware
//
#include <hip/hip_runtime.h>
#include <hip/hip_bf16.h>

// MI455X / gfx1250 fused attention:
//   K1: Wq/Wk/Wv f32 -> f16
//   K2: QKV projection via v_wmma_f32_16x16x32_f16. X staged f16 in LDS and
//       re-read per k-step (not register-pinned); 4 accumulator chains per
//       task; software pipeline enforced with sched_group_barrier (2 k-steps
//       of operand loads kept in flight ahead of the WMMAs).
//   K3: flash attention, 1 WG (4 waves) per (batch, 16-query tile):
//       S=QK^T d-split across waves + LDS reduction, online softmax via
//       half-wave shuffles, P through LDS, O accumulation e-split per wave.
//       Q staged in LDS; S- and O-loops pipelined with sched_group_barrier.
// Workspace: 2MB (Wh) + 32MB (Qh) + 32MB (Kh) + 32MB (Vt) = ~98MB.

#define ADIM 512

typedef _Float16 v4h  __attribute__((ext_vector_type(4)));
typedef _Float16 v8h  __attribute__((ext_vector_type(8)));
typedef _Float16 v16h __attribute__((ext_vector_type(16)));
typedef float    v8f  __attribute__((ext_vector_type(8)));
typedef float    v4f  __attribute__((ext_vector_type(4)));

// Scheduling-group masks (AMDGPU sched_group_barrier)
#define M_WMMA    0x008   // MFMA/WMMA
#define M_VMEM_RD 0x020   // VMEM read
#define M_DS_RD   0x100   // DS read

#if defined(__has_builtin)
#  if __has_builtin(__builtin_amdgcn_sched_group_barrier)
#    define SGB(m, s) __builtin_amdgcn_sched_group_barrier((m), (s), 0)
#  endif
#endif
#ifndef SGB
#  define SGB(m, s)
#endif

static __device__ __forceinline__ v16h cat8(v8h lo, v8h hi) {
  return __builtin_shufflevector(lo, hi, 0,1,2,3,4,5,6,7,8,9,10,11,12,13,14,15);
}

static __device__ __forceinline__ v8f wmma_f16(v16h a, v16h b, v8f c) {
  // D = A(16x32 f16) * B(32x16 f16) + C(16x16 f32)
  return __builtin_amdgcn_wmma_f32_16x16x32_f16(
      /*neg_a=*/false, a, /*neg_b=*/false, b,
      /*c_mod=*/(short)0, c, /*reuse_a=*/false, /*reuse_b=*/false);
}

static __device__ __forceinline__ v16h ld32h(const _Float16* p) {
  return cat8(*(const v8h*)p, *(const v8h*)(p + 8));
}

// ---------------------------------------------------------------- K1: weights
__global__ __launch_bounds__(256) void wconv_kernel(
    const float* __restrict__ Wq, const float* __restrict__ Wk,
    const float* __restrict__ Wv, _Float16* __restrict__ Wh) {
  const int per = ADIM * ADIM;
  int i = blockIdx.x * 256 + threadIdx.x;     // 0 .. 3*per-1
  if (i < per)            Wh[i] = (_Float16)Wq[i];
  else if (i < 2 * per)   Wh[i] = (_Float16)Wk[i - per];
  else                    Wh[i] = (_Float16)Wv[i - 2 * per];
}

// ------------------------------------------------------------ K2: QKV project
// grid = 2048 (16-row tiles of [B*N, 512]), block = 128 (4 waves)
__global__ __launch_bounds__(128, 1) void qkv_kernel(
    const float* __restrict__ emb, const _Float16* __restrict__ Wh,
    _Float16* __restrict__ Qh, _Float16* __restrict__ Kh,
    _Float16* __restrict__ Vt) {
  __shared__ __align__(16) _Float16 Xh[16][ADIM];   // 16KB

  const int tid  = threadIdx.x;
  const size_t rbase = (size_t)blockIdx.x * 16;

  // Cooperative load of X tile, f32 -> f16 into LDS.
#pragma unroll
  for (int i = 0; i < 16; ++i) {
    int f = tid * 64 + i * 4;                 // 0..8191, multiple of 4
    int r = f >> 9, d = f & 511;
    const float4 v = *(const float4*)(emb + (rbase + r) * ADIM + d);
    v4h h; h[0] = (_Float16)v.x; h[1] = (_Float16)v.y;
           h[2] = (_Float16)v.z; h[3] = (_Float16)v.w;
    *(v4h*)&Xh[r][d] = h;
  }
  __syncthreads();

  const int w    = tid >> 5;
  const int lane = tid & 31;
  const int n    = lane & 15;
  const int half = lane >> 4;

  // Each wave: 6 tasks x 4 e-chunks (96 e-chunk tasks total across 4 waves).
  for (int t = 0; t < 6; ++t) {
    // Keep LDS A-loads inside the loop (defeat LICM re-hoisting).
    asm volatile("" ::: "memory");

    const int q   = w * 6 + t;                // 0..23, uniform per wave
    const int mat = q >> 3;                   // 0=Q 1=K 2=V
    const int u   = q & 7;                    // quad index
    const _Float16* Wbase = Wh + (size_t)mat * ADIM * ADIM;
    const _Float16* Wrow[4];
#pragma unroll
    for (int r = 0; r < 4; ++r)
      Wrow[r] = Wbase + (size_t)((u * 4 + r) * 16 + n) * ADIM + (half ? 16 : 0);

    v8f c[4] = {};
#pragma unroll
    for (int dc = 0; dc < 16; ++dc) {
      const int d0 = dc * 32;
      // A fragment from LDS (2 x ds_load_b128)
      v16h a = cat8(*(const v8h*)&Xh[n][d0 + (half ? 16 : 0)],
                    *(const v8h*)&Xh[n][d0 + 8 + (half ? 16 : 0)]);
#pragma unroll
      for (int r = 0; r < 4; ++r)
        c[r] = wmma_f16(a, ld32h(Wrow[r] + d0), c[r]);
    }

    // Software pipeline: 2 k-steps of operand loads in flight ahead of WMMAs.
    SGB(M_DS_RD, 2); SGB(M_VMEM_RD, 8);
    SGB(M_DS_RD, 2); SGB(M_VMEM_RD, 8);
#pragma unroll
    for (int i = 0; i < 14; ++i) {
      SGB(M_WMMA, 4); SGB(M_DS_RD, 2); SGB(M_VMEM_RD, 8);
    }
    SGB(M_WMMA, 4); SGB(M_WMMA, 4);

    if (mat == 2) {
      // V stored transposed: Vt[b][e][k]
#pragma unroll
      for (int r = 0; r < 4; ++r) {
        const int e = (u * 4 + r) * 16 + n;
#pragma unroll
        for (int j = 0; j < 8; ++j) {
          size_t row = rbase + j + 8 * half;  // global flat row = b*1024 + k
          int bb = (int)(row >> 10), nk = (int)(row & 1023);
          Vt[((size_t)bb * ADIM + e) * 1024 + nk] = (_Float16)c[r][j];
        }
      }
    } else {
      _Float16* dst = (mat == 0) ? Qh : Kh;
#pragma unroll
      for (int r = 0; r < 4; ++r) {
        const int e = (u * 4 + r) * 16 + n;
#pragma unroll
        for (int j = 0; j < 8; ++j) {
          size_t row = rbase + j + 8 * half;
          dst[row * ADIM + e] = (_Float16)c[r][j];
        }
      }
    }
  }
}

// --------------------------------------------------------- K3: flash attention
// grid = 2048 (= B(32) * 64 q-tiles), block = 128 (4 waves).
// Wave w: d-range [w*128, w*128+128) for S partials, e-range [w*128, ...) for O.
__global__ __launch_bounds__(128, 1) void attn_kernel(
    const _Float16* __restrict__ Qh, const _Float16* __restrict__ Kh,
    const _Float16* __restrict__ Vt, float* __restrict__ out) {
  __shared__ __align__(16) _Float16 QT[16][ADIM];        // 16KB: Q tile
  __shared__ __align__(16) float     Spart[16][32][4];   // 8KB: [slot][lane][wave]
  __shared__ __align__(16) _Float16  Ptile[4][16][32];   // 4KB: per-wave P tile

  const int tid  = threadIdx.x;
  const int w    = tid >> 5;
  const int lane = tid & 31;
  const int n    = lane & 15;
  const int half = lane >> 4;

  const int wg   = blockIdx.x;
  const int b    = wg >> 6;
  const size_t rbase = (size_t)wg * 16;       // global flat query row
  const int dbase = w * 128;
  const int eoff  = w * 128;
  const float sc  = 0.044194173824159216f;    // 1/sqrt(512)

  const _Float16* Kb = Kh + (size_t)b * 1024 * ADIM;
  const _Float16* Vb = Vt + (size_t)b * ADIM * 1024;

  // Stage Q tile (16 x 512 f16 = 16KB) into LDS cooperatively.
#pragma unroll
  for (int i = 0; i < 8; ++i) {
    int f = tid * 64 + i * 8;                 // halfs, 0..8191
    *(v8h*)((_Float16*)QT + f) = *(const v8h*)(Qh + rbase * ADIM + f);
  }
  __syncthreads();

  v8f acc[8] = {};
  float mrow[8], lrow[8];
#pragma unroll
  for (int j = 0; j < 8; ++j) { mrow[j] = -1.0e30f; lrow[j] = 0.0f; }

  for (int kt = 0; kt < 32; ++kt) {
    const int kbase = kt * 32;

    // Prefetch next key tile (gfx1250 global_prefetch_b8).
    if (kt < 31) {
      const _Float16* pf = Kb + (size_t)(kbase + 32 + n) * ADIM
                              + dbase + (half ? 16 : 0);
      __builtin_prefetch(pf, 0, 3);
      __builtin_prefetch(pf + 16 * ADIM, 0, 3);
    }

    // ---- partial S = Q * K^T over this wave's 128 d-dims (2 key columns)
    const _Float16* k0 = Kb + (size_t)(kbase + n) * ADIM + dbase + (half ? 16 : 0);
    const _Float16* k1 = Kb + (size_t)(kbase + 16 + n) * ADIM + dbase + (half ? 16 : 0);
    v8f s0 = {}, s1 = {};
#pragma unroll
    for (int dc = 0; dc < 4; ++dc) {
      const int d0 = dbase + dc * 32;
      v16h qa = cat8(*(const v8h*)&QT[n][d0 + (half ? 16 : 0)],
                     *(const v8h*)&QT[n][d0 + 8 + (half ? 16 : 0)]);
      s0 = wmma_f16(qa, ld32h(k0 + dc * 32), s0);
      s1 = wmma_f16(qa, ld32h(k1 + dc * 32), s1);
    }
    // Pipeline: 1 k-step of loads ahead of the WMMA pairs.
    SGB(M_DS_RD, 2); SGB(M_VMEM_RD, 4);
#pragma unroll
    for (int i = 0; i < 3; ++i) {
      SGB(M_WMMA, 2); SGB(M_DS_RD, 2); SGB(M_VMEM_RD, 4);
    }
    SGB(M_WMMA, 2);

    // ---- reduce the 4 d-partials through LDS
#pragma unroll
    for (int j = 0; j < 8; ++j) {
      Spart[j][lane][w]     = s0[j];
      Spart[8 + j][lane][w] = s1[j];
    }
    __syncthreads();
#pragma unroll
    for (int j = 0; j < 8; ++j) {
      v4f p0 = *(const v4f*)&Spart[j][lane][0];
      v4f p1 = *(const v4f*)&Spart[8 + j][lane][0];
      s0[j] = ((p0[0] + p0[1]) + (p0[2] + p0[3])) * sc;
      s1[j] = ((p1[0] + p1[1]) + (p1[2] + p1[3])) * sc;
    }
    __syncthreads();   // all reads done before next iteration's writes

    // ---- online softmax (per row; rows live in lane-halves, half-wave shuffles)
#pragma unroll
    for (int j = 0; j < 8; ++j) {
      float t = fmaxf(s0[j], s1[j]);
      t = fmaxf(t, __shfl_xor(t, 1, 32));
      t = fmaxf(t, __shfl_xor(t, 2, 32));
      t = fmaxf(t, __shfl_xor(t, 4, 32));
      t = fmaxf(t, __shfl_xor(t, 8, 32));
      const float mnew = fmaxf(mrow[j], t);
      const float e0 = __expf(s0[j] - mnew);
      const float e1 = __expf(s1[j] - mnew);
      float rs = e0 + e1;
      rs += __shfl_xor(rs, 1, 32);
      rs += __shfl_xor(rs, 2, 32);
      rs += __shfl_xor(rs, 4, 32);
      rs += __shfl_xor(rs, 8, 32);
      const float scale = __expf(mrow[j] - mnew);
      lrow[j] = lrow[j] * scale + rs;
      mrow[j] = mnew;
#pragma unroll
      for (int ec = 0; ec < 8; ++ec) acc[ec][j] *= scale;
      // store P (this wave's private region; same-wave DS ordering suffices)
      Ptile[w][j + 8 * half][n]      = (_Float16)e0;
      Ptile[w][j + 8 * half][16 + n] = (_Float16)e1;
    }

    // ---- read P back as an A-fragment (lane = row n)
    const _Float16* pp = &Ptile[w][n][half ? 16 : 0];
    v16h pa = cat8(*(const v8h*)pp, *(const v8h*)(pp + 8));

    // ---- O += P * V over this wave's 128 e-dims (independent chains)
    const _Float16* vbase = Vb + (size_t)(eoff + n) * 1024 + kbase + (half ? 16 : 0);
#pragma unroll
    for (int ec = 0; ec < 8; ++ec) {
      v16h vbf = ld32h(vbase + (size_t)ec * 16 * 1024);
      acc[ec] = wmma_f16(pa, vbf, acc[ec]);
    }
    // Pipeline: 2 V fragments in flight ahead of the WMMAs.
    SGB(M_VMEM_RD, 2); SGB(M_VMEM_RD, 2);
#pragma unroll
    for (int i = 0; i < 6; ++i) {
      SGB(M_WMMA, 1); SGB(M_VMEM_RD, 2);
    }
    SGB(M_WMMA, 1); SGB(M_WMMA, 1);
  }

  // ---- epilogue: divide by softmax denominator, store f32
#pragma unroll
  for (int j = 0; j < 8; ++j) {
    const float inv = 1.0f / lrow[j];
    const size_t row = rbase + j + 8 * half;
#pragma unroll
    for (int ec = 0; ec < 8; ++ec) {
      out[row * ADIM + eoff + ec * 16 + n] = acc[ec][j] * inv;
    }
  }
}

// ------------------------------------------------------------------- launcher
extern "C" void kernel_launch(void* const* d_in, const int* in_sizes, int n_in,
                              void* d_out, int out_size, void* d_ws, size_t ws_size,
                              hipStream_t stream) {
  (void)in_sizes; (void)n_in; (void)out_size; (void)ws_size;
  const float* emb = (const float*)d_in[0];
  const float* Wq  = (const float*)d_in[1];
  const float* Wk  = (const float*)d_in[2];
  const float* Wv  = (const float*)d_in[3];
  float* out = (float*)d_out;

  char* ws = (char*)d_ws;
  const size_t QH_OFF   = 2u << 20;                          // 2MB (Wh region)
  const size_t QK_BYTES = (size_t)32 * 1024 * ADIM * 2;      // 32MB each
  _Float16* Wh = (_Float16*)ws;
  _Float16* Qh = (_Float16*)(ws + QH_OFF);
  _Float16* Kh = (_Float16*)(ws + QH_OFF + QK_BYTES);
  _Float16* Vt = (_Float16*)(ws + QH_OFF + 2 * QK_BYTES);

  wconv_kernel<<<(3 * ADIM * ADIM) / 256, 256, 0, stream>>>(Wq, Wk, Wv, Wh);
  qkv_kernel<<<2048, 128, 0, stream>>>(emb, Wh, Qh, Kh, Vt);
  attn_kernel<<<2048, 128, 0, stream>>>(Qh, Kh, Vt, out);
}
